// EncoderLayer_52905407152255
// MI455X (gfx1250) — matrix operand order repaired and checked by reading, not compiled, not run on hardware
//
#include <hip/hip_runtime.h>
#include <hip/hip_bf16.h>
#include <math.h>

typedef __attribute__((ext_vector_type(16))) __bf16        v16bf;
typedef __attribute__((ext_vector_type(8)))  float         v8f;
typedef __attribute__((ext_vector_type(4)))  unsigned int  v4u;

#define NEG_SLOPE 0.2f

// ---------------------------------------------------------------------------
// fp32 -> bf16 conversion (used once per weight matrix, <=16K elements)
// ---------------------------------------------------------------------------
__global__ void k_cvt_bf16(const float* __restrict__ in, __bf16* __restrict__ out, int n) {
    int i = blockIdx.x * blockDim.x + threadIdx.x;
    if (i < n) out[i] = (__bf16)in[i];
}

// ---------------------------------------------------------------------------
// WMMA GEMM:  C[M,128] = A[M,K] @ W[128,K]^T
// A fp32 in memory (converted per k-step, amortized over 8 column tiles);
// W pre-converted bf16, row-major [128,K].
// One wave computes a 16x128 tile; K in {64,128}.
// A frag (16x32): lane l holds row (l&15); 16 contiguous K {b..b+15},
//   b = (l>>4)*16.   B frag (32x16): lane l holds col (l&15) = W row,
//   16 contiguous bf16 at K offset (l>>4)*16 -> two 16B raw loads, no VALU.
// ---------------------------------------------------------------------------
union BFrag { v16bf v; v4u u[2]; };

__global__ void k_gemm_wmma(const float* __restrict__ A, const __bf16* __restrict__ Wb,
                            float* __restrict__ C, int M, int K) {
    int wid  = (blockIdx.x * blockDim.x + threadIdx.x) >> 5;
    int lane = threadIdx.x & 31;
    int row0 = wid << 4;
    if (row0 >= M) return;
    int lmod = lane & 15;
    int lhi  = lane >> 4;
    int arow = row0 + lmod; if (arow >= M) arow = M - 1;
    const float* Arow = A + (size_t)arow * K;

    v8f acc[8] = {};
    for (int k0 = 0; k0 < K; k0 += 32) {
        // ---- A fragment (fp32 loads + packed cvt, once per k-step)
        int ka = k0 + (lhi << 4);
        v16bf a;
#pragma unroll
        for (int j = 0; j < 8; ++j) {
            a[j]     = (__bf16)Arow[ka + j];
            a[j + 8] = (__bf16)Arow[ka + 8 + j];
        }
        int kb = k0 + (lhi << 4);
#pragma unroll
        for (int nt = 0; nt < 8; ++nt) {
            const __bf16* Wp = Wb + (size_t)(nt * 16 + lmod) * K + kb;
            BFrag b;
            b.u[0] = ((const v4u*)Wp)[0];   // 8 bf16
            b.u[1] = ((const v4u*)Wp)[1];   // 8 bf16
            acc[nt] = __builtin_amdgcn_wmma_f32_16x16x32_bf16(
                false, a, false, b.v, (short)0, acc[nt], false, false);
        }
    }
    // ---- store: element v of acc -> row (row0 + lhi*8 + v), col (nt*16 + lmod)
    int rbase = row0 + (lhi << 3);
#pragma unroll
    for (int nt = 0; nt < 8; ++nt) {
        float* Cp = C + (size_t)rbase * 128 + nt * 16 + lmod;
#pragma unroll
        for (int v = 0; v < 8; ++v) {
            int m = rbase + v;
            if (m < M) Cp[(size_t)v * 128] = acc[nt][v];
        }
    }
}

// ---------------------------------------------------------------------------
// LayerNorm over 128 channels, one wave per row (lane owns 4 channels).
// ---------------------------------------------------------------------------
__device__ inline float wave_sum(float x) {
#pragma unroll
    for (int o = 16; o > 0; o >>= 1) x += __shfl_xor(x, o);
    return x;
}

__global__ void k_ln(const float* __restrict__ x, const float* __restrict__ g,
                     const float* __restrict__ b, float* __restrict__ o, int M) {
    int row = (blockIdx.x * blockDim.x + threadIdx.x) >> 5;
    int lane = threadIdx.x & 31;
    if (row >= M) return;
    float4 v = ((const float4*)(x + (size_t)row * 128))[lane];
    float mu = wave_sum(v.x + v.y + v.z + v.w) * (1.0f / 128.0f);
    float dx = v.x - mu, dy = v.y - mu, dz = v.z - mu, dw = v.w - mu;
    float var = wave_sum(dx * dx + dy * dy + dz * dz + dw * dw) * (1.0f / 128.0f);
    float rstd = rsqrtf(var + 1e-5f);
    float4 gg = ((const float4*)g)[lane];
    float4 bb = ((const float4*)b)[lane];
    float4 r;
    r.x = dx * rstd * gg.x + bb.x;
    r.y = dy * rstd * gg.y + bb.y;
    r.z = dz * rstd * gg.z + bb.z;
    r.w = dw * rstd * gg.w + bb.w;
    ((float4*)(o + (size_t)row * 128))[lane] = r;
}

// ---------------------------------------------------------------------------
// Attention projections: outA = x @ attA, outB = x @ attB  ([N,128]@[128,4])
// ---------------------------------------------------------------------------
__global__ void k_attvec(const float* __restrict__ x,
                         const float* __restrict__ attA, const float* __restrict__ attB,
                         float* __restrict__ outA, float* __restrict__ outB, int M) {
    int row = (blockIdx.x * blockDim.x + threadIdx.x) >> 5;
    int lane = threadIdx.x & 31;
    if (row >= M) return;
    float4 v = ((const float4*)(x + (size_t)row * 128))[lane];
    float xv[4] = {v.x, v.y, v.z, v.w};
    int c0 = lane * 4;
    float sa[4] = {0, 0, 0, 0}, sb[4] = {0, 0, 0, 0};
#pragma unroll
    for (int j = 0; j < 4; ++j)
#pragma unroll
        for (int h = 0; h < 4; ++h) {
            sa[h] += xv[j] * attA[(size_t)(c0 + j) * 4 + h];
            sb[h] += xv[j] * attB[(size_t)(c0 + j) * 4 + h];
        }
#pragma unroll
    for (int h = 0; h < 4; ++h) { sa[h] = wave_sum(sa[h]); sb[h] = wave_sum(sb[h]); }
    if (lane == 0) {
#pragma unroll
        for (int h = 0; h < 4; ++h) {
            outA[(size_t)row * 4 + h] = sa[h];
            outB[(size_t)row * 4 + h] = sb[h];
        }
    }
}

// ---------------------------------------------------------------------------
// Atomic float max via int/uint monotonic trick (init must be negative float).
// ---------------------------------------------------------------------------
__device__ inline void atomicMaxF(float* a, float v) {
    if (v >= 0.0f) atomicMax((int*)a, __float_as_int(v));
    else           atomicMin((unsigned int*)a, __float_as_uint(v));
}

__device__ inline float lrelu(float x) { return x > 0.0f ? x : NEG_SLOPE * x; }

// score[e,h] = leaky_relu(aDst[dst[e],h] + bSrc[src[e],h]); segment max by dst.
// Edges e>=E are self loops (src=dst=e-E).
__global__ void k_score(const int* __restrict__ src, const int* __restrict__ dst,
                        int E, int nSelf,
                        const float* __restrict__ aD, const float* __restrict__ bS,
                        float* __restrict__ sc, float* __restrict__ smax) {
    int e = blockIdx.x * blockDim.x + threadIdx.x;
    if (e >= E + nSelf) return;
    int s_, d_;
    if (e < E) { s_ = src[e]; d_ = dst[e]; } else { s_ = d_ = e - E; }
    float4 a = ((const float4*)aD)[d_];
    float4 b = ((const float4*)bS)[s_];
    float4 r;
    r.x = lrelu(a.x + b.x); r.y = lrelu(a.y + b.y);
    r.z = lrelu(a.z + b.z); r.w = lrelu(a.w + b.w);
    ((float4*)sc)[e] = r;
    float* m = smax + (size_t)d_ * 4;
    atomicMaxF(m + 0, r.x); atomicMaxF(m + 1, r.y);
    atomicMaxF(m + 2, r.z); atomicMaxF(m + 3, r.w);
}

// sc[e,h] = exp(sc[e,h] - smax[dst]); den[dst,h] += sc[e,h]
__global__ void k_expden(const int* __restrict__ dst, int E, int nSelf,
                         float* __restrict__ sc, const float* __restrict__ smax,
                         float* __restrict__ den) {
    int e = blockIdx.x * blockDim.x + threadIdx.x;
    if (e >= E + nSelf) return;
    int d_ = (e < E) ? dst[e] : e - E;
    float4 s = ((float4*)sc)[e];
    float4 m = ((const float4*)smax)[d_];
    s.x = __expf(s.x - m.x); s.y = __expf(s.y - m.y);
    s.z = __expf(s.z - m.z); s.w = __expf(s.w - m.w);
    ((float4*)sc)[e] = s;
    float* d = den + (size_t)d_ * 4;
    atomicAdd(d + 0, s.x); atomicAdd(d + 1, s.y);
    atomicAdd(d + 2, s.z); atomicAdd(d + 3, s.w);
}

// out[dst] += scale * mean_h(sc[e,h]/(den[dst,h]+1e-16)) * feats[src]
// One wave per edge; lane owns 4 channels -> coalesced 512B gathers.
__global__ void k_agg(const int* __restrict__ src, const int* __restrict__ dst,
                      int E, int nSelf,
                      const float* __restrict__ sc, const float* __restrict__ den,
                      const float* __restrict__ feats, float* __restrict__ out,
                      const float* __restrict__ pw, int pidx) {
    int t = blockIdx.x * blockDim.x + threadIdx.x;
    int lane = t & 31;
    int e = t >> 5;
    if (e >= E + nSelf) return;
    int s_, d_;
    if (e < E) { s_ = src[e]; d_ = dst[e]; } else { s_ = d_ = e - E; }
    __builtin_prefetch(feats + (size_t)s_ * 128 + lane * 4, 0, 0);
    float4 al = ((const float4*)sc)[e];
    float4 dn = ((const float4*)den)[d_];
    float ab = 0.25f * (al.x / (dn.x + 1e-16f) + al.y / (dn.y + 1e-16f) +
                        al.z / (dn.z + 1e-16f) + al.w / (dn.w + 1e-16f));
    if (pw) ab *= pw[pidx];
    float4 f = ((const float4*)(feats + (size_t)s_ * 128))[lane];
    float* op = out + (size_t)d_ * 128 + lane * 4;
    atomicAdd(op + 0, ab * f.x);
    atomicAdd(op + 1, ab * f.y);
    atomicAdd(op + 2, ab * f.z);
    atomicAdd(op + 3, ab * f.w);
}

__global__ void k_fill(float* __restrict__ p, float v, size_t n) {
    size_t i = (size_t)blockIdx.x * blockDim.x + threadIdx.x;
    if (i < n) p[i] = v;
}

// x += acc + pw[0]*bias[0,:] + pw[1]*bias[1,:]
__global__ void k_resid(float* __restrict__ x, const float* __restrict__ acc,
                        const float* __restrict__ pw, const float* __restrict__ bias, int M) {
    size_t i = (size_t)blockIdx.x * blockDim.x + threadIdx.x;
    if (i >= (size_t)M * 128) return;
    int c = (int)(i & 127);
    x[i] += acc[i] + pw[0] * bias[c] + pw[1] * bias[128 + c];
}

// out[*, i, c] += b1[c] + b2[c]  over both stacked halves
__global__ void k_outbias(float* __restrict__ out, const float* __restrict__ b1,
                          const float* __restrict__ b2, int M) {
    size_t i = (size_t)blockIdx.x * blockDim.x + threadIdx.x;
    if (i >= (size_t)2 * M * 128) return;
    int c = (int)(i & 127);
    out[i] += b1[c] + b2[c];
}

// ---------------------------------------------------------------------------
// Host orchestration
// ---------------------------------------------------------------------------
static inline int cdiv(long a, long b) { return (int)((a + b - 1) / b); }

extern "C" void kernel_launch(void* const* d_in, const int* in_sizes, int n_in,
                              void* d_out, int out_size, void* d_ws, size_t ws_size,
                              hipStream_t stream) {
    const float* x_var     = (const float*)d_in[0];
    const float* x_cls     = (const float*)d_in[1];
    const int*   mp_var    = (const int*)d_in[2];
    const int*   mp_cls    = (const int*)d_in[3];
    const int*   adj_pos   = (const int*)d_in[4];
    const int*   adj_neg   = (const int*)d_in[5];
    const float* W_var_emb = (const float*)d_in[6];
    const float* W_cls_emb = (const float*)d_in[7];
    const float* ln_var_g  = (const float*)d_in[8];
    const float* ln_var_b  = (const float*)d_in[9];
    const float* ln_cls_g  = (const float*)d_in[10];
    const float* ln_cls_b  = (const float*)d_in[11];
    const float* var_pw    = (const float*)d_in[12];
    const float* cls_pw    = (const float*)d_in[13];
    const float* sv_W      = (const float*)d_in[14];
    const float* sv_al     = (const float*)d_in[15];
    const float* sv_ar     = (const float*)d_in[16];
    const float* sv_b      = (const float*)d_in[17];
    const float* sc_W      = (const float*)d_in[18];
    const float* sc_al     = (const float*)d_in[19];
    const float* sc_ar     = (const float*)d_in[20];
    const float* sc_b      = (const float*)d_in[21];
    const float* cp_Wl     = (const float*)d_in[22];
    const float* cp_Wr     = (const float*)d_in[23];
    const float* cp_al     = (const float*)d_in[24];
    const float* cp_ar     = (const float*)d_in[25];
    const float* cp_b      = (const float*)d_in[26];
    const float* cn_Wl     = (const float*)d_in[27];
    const float* cn_Wr     = (const float*)d_in[28];
    const float* cn_al     = (const float*)d_in[29];
    const float* cn_ar     = (const float*)d_in[30];
    const float* cn_b      = (const float*)d_in[31];

    const int N = in_sizes[0] / 64;   // 50000
    const int E = in_sizes[4] / 2;    // 640000
    const size_t NF = (size_t)N * 128;

    float* W0   = (float*)d_ws;
    float* XV   = W0;                 // [N,128] embedded var (residual stream)
    float* XC   = W0 + NF;            // [N,128] embedded cls
    float* HV   = W0 + 2 * NF;        // LN(xv); later cross xl_var
    float* HC   = W0 + 3 * NF;        // LN(xc); later cross xr_cls
    float* XL   = W0 + 4 * NF;        // per-path transformed feats
    float* ACV  = W0 + 5 * NF;        // self-attn accumulator (var)
    float* ACC  = W0 + 6 * NF;        // self-attn accumulator (cls)
    float* AL   = W0 + 7 * NF;
    float* AR   = AL + (size_t)N * 4;
    float* AL2  = AR + (size_t)N * 4;
    float* AR2  = AL2 + (size_t)N * 4;
    float* SMAX = AR2 + (size_t)N * 4;
    float* DEN  = SMAX + (size_t)N * 4;
    float* SC   = DEN + (size_t)N * 4;            // [(E+N),4] edge scores
    __bf16* WBF = (__bf16*)(SC + (size_t)(E + N) * 4); // bf16 weight staging (32KB)

    const int gemmGrid = cdiv((long)cdiv(N, 16) * 32, 256);
    const int rowGrid  = cdiv((long)N * 32, 256);
    const int nfGrid   = cdiv((long)N * 128, 256);
    const int n4Grid   = cdiv((long)N * 4, 256);

    // GEMM helper: stage weights (128xK fp32) as bf16, then WMMA GEMM
    auto gemm = [&](const float* Am, const float* Wm, float* Cm, int K) {
        int wn = 128 * K;
        k_cvt_bf16<<<cdiv(wn, 256), 256, 0, stream>>>(Wm, WBF, wn);
        k_gemm_wmma<<<gemmGrid, 256, 0, stream>>>(Am, WBF, Cm, N, K);
    };

    // ---- embeddings + LN ----
    gemm(x_var, W_var_emb, XV, 64);
    gemm(x_cls, W_cls_emb, XC, 64);
    k_ln<<<rowGrid, 256, 0, stream>>>(XV, ln_var_g, ln_var_b, HV, N);
    k_ln<<<rowGrid, 256, 0, stream>>>(XC, ln_cls_g, ln_cls_b, HC, N);

    // ---- self attention (per side, per meta-path), heads collapsed pre-scatter
    auto self_side = [&](const float* Hm, const int* mp, const float* sW,
                         const float* sal, const float* sar, const float* pwp,
                         float* ACCbuf) {
        k_fill<<<nfGrid, 256, 0, stream>>>(ACCbuf, 0.0f, NF);
        for (int p = 0; p < 2; ++p) {
            gemm(Hm, sW + (size_t)p * 128 * 128, XL, 128);
            k_attvec<<<rowGrid, 256, 0, stream>>>(XL, sal + (size_t)p * 512,
                                                  sar + (size_t)p * 512, AL, AR, N);
            k_fill<<<n4Grid, 256, 0, stream>>>(SMAX, -INFINITY, (size_t)N * 4);
            k_fill<<<n4Grid, 256, 0, stream>>>(DEN, 0.0f, (size_t)N * 4);
            const int* src = mp + (size_t)p * 2 * E;
            const int* dst = src + E;
            const int tot = E + N;
            k_score<<<cdiv(tot, 256), 256, 0, stream>>>(src, dst, E, N, AL, AR, SC, SMAX);
            k_expden<<<cdiv(tot, 256), 256, 0, stream>>>(dst, E, N, SC, SMAX, DEN);
            k_agg<<<cdiv((long)tot * 32, 256), 256, 0, stream>>>(src, dst, E, N, SC, DEN,
                                                                 XL, ACCbuf, pwp, p);
        }
    };
    self_side(HV, mp_var, sv_W, sv_al, sv_ar, var_pw, ACV);
    self_side(HC, mp_cls, sc_W, sc_al, sc_ar, cls_pw, ACC);
    k_resid<<<nfGrid, 256, 0, stream>>>(XV, ACV, var_pw, sv_b, N);
    k_resid<<<nfGrid, 256, 0, stream>>>(XC, ACC, cls_pw, sc_b, N);

    // ---- cross (bipartite) attention over pos / neg literal edges ----
    float* out = (float*)d_out;
    k_fill<<<cdiv((long)2 * N * 128, 256), 256, 0, stream>>>(out, 0.0f, (size_t)2 * N * 128);

    auto cross = [&](const int* adj, const float* Wl, const float* Wr,
                     const float* cal, const float* car) {
        float* XLV = HV;  // reuse LN buffers as cross-transformed feats
        float* XRC = HC;
        gemm(XV, Wl, XLV, 128);
        gemm(XC, Wr, XRC, 128);
        // xl@att_l -> al, xl@att_r -> ar_ ; xr@att_l -> al_, xr@att_r -> ar
        k_attvec<<<rowGrid, 256, 0, stream>>>(XLV, cal, car, AL, AR2, N);
        k_attvec<<<rowGrid, 256, 0, stream>>>(XRC, cal, car, AL2, AR, N);
        const int* cidx = adj;      // clause side
        const int* vidx = adj + E;  // var side
        // direction -> var nodes: score = al[v] + ar[c], segmented by v
        k_fill<<<n4Grid, 256, 0, stream>>>(SMAX, -INFINITY, (size_t)N * 4);
        k_fill<<<n4Grid, 256, 0, stream>>>(DEN, 0.0f, (size_t)N * 4);
        k_score<<<cdiv(E, 256), 256, 0, stream>>>(cidx, vidx, E, 0, AL, AR, SC, SMAX);
        k_expden<<<cdiv(E, 256), 256, 0, stream>>>(vidx, E, 0, SC, SMAX, DEN);
        k_agg<<<cdiv((long)E * 32, 256), 256, 0, stream>>>(cidx, vidx, E, 0, SC, DEN,
                                                           XRC, out, nullptr, 0);
        // direction -> cls nodes: score_ = ar_[v] + al_[c], segmented by c
        k_fill<<<n4Grid, 256, 0, stream>>>(SMAX, -INFINITY, (size_t)N * 4);
        k_fill<<<n4Grid, 256, 0, stream>>>(DEN, 0.0f, (size_t)N * 4);
        k_score<<<cdiv(E, 256), 256, 0, stream>>>(vidx, cidx, E, 0, AL2, AR2, SC, SMAX);
        k_expden<<<cdiv(E, 256), 256, 0, stream>>>(cidx, E, 0, SC, SMAX, DEN);
        k_agg<<<cdiv((long)E * 32, 256), 256, 0, stream>>>(vidx, cidx, E, 0, SC, DEN,
                                                           XLV, out + NF, nullptr, 0);
    };
    cross(adj_pos, cp_Wl, cp_Wr, cp_al, cp_ar);
    cross(adj_neg, cn_Wl, cn_Wr, cn_al, cn_ar);

    k_outbias<<<cdiv((long)2 * N * 128, 256), 256, 0, stream>>>(out, cp_b, cn_b, N);
}
